// TransformerBlock_18554258719073
// MI455X (gfx1250) — matrix operand order repaired and checked by reading, not compiled, not run on hardware
//
#include <hip/hip_runtime.h>
#include <cstdint>

#define EMB 1024
#define HEADS 16
#define SEQ 2048
#define DFF 4096
#define HD 64

typedef __bf16 bf16_t;
typedef __bf16 v16bf __attribute__((ext_vector_type(16)));
typedef float  v8f   __attribute__((ext_vector_type(8)));

union Frag16 { uint4 u[2]; v16bf v; };

__device__ inline v8f wmma_bf16(v16bf a, v16bf b, v8f c) {
#if defined(__HIP_DEVICE_COMPILE__)
  return __builtin_amdgcn_wmma_f32_16x16x32_bf16(false, a, false, b, (short)0, c,
                                                 false, false);
#else
  (void)a; (void)b;
  return c;
#endif
}

__device__ inline v8f v8f_zero() {
  v8f z;
#pragma unroll
  for (int i = 0; i < 8; ++i) z[i] = 0.f;
  return z;
}

// CDNA5 async DMA: global -> LDS, tracked by ASYNCcnt (CDNA5 ISA)
__device__ inline void async_ld_b128(uint32_t lds_addr, const void* gaddr) {
  asm volatile("global_load_async_to_lds_b128 %0, %1, off"
               :
               : "v"(lds_addr), "v"((unsigned long long)(uintptr_t)gaddr)
               : "memory");
}
__device__ inline void wait_async0() {
  asm volatile("s_wait_asynccnt 0x0" ::: "memory");
}

// ---------------------------------------------------------------- f32 -> bf16
__global__ void cvt_f32_bf16(const float* __restrict__ in, bf16_t* __restrict__ out, int n) {
  int i = (blockIdx.x * blockDim.x + threadIdx.x) * 4;
  if (i + 3 < n) {
    float4 v = *(const float4*)(in + i);
    out[i + 0] = (bf16_t)v.x;
    out[i + 1] = (bf16_t)v.y;
    out[i + 2] = (bf16_t)v.z;
    out[i + 3] = (bf16_t)v.w;
  }
}

// ---------------------------------------------------------------- LayerNorm
__global__ __launch_bounds__(256) void layernorm_bf16(const float* __restrict__ x,
                                                      const float* __restrict__ gw,
                                                      const float* __restrict__ gb,
                                                      bf16_t* __restrict__ out) {
  __shared__ float rs[256], rq[256];
  const int row = blockIdx.x, tid = threadIdx.x;
  const float* xr = x + (size_t)row * EMB;
  float4 v = *(const float4*)(xr + tid * 4);
  rs[tid] = v.x + v.y + v.z + v.w;
  rq[tid] = v.x * v.x + v.y * v.y + v.z * v.z + v.w * v.w;
  __syncthreads();
  for (int st = 128; st > 0; st >>= 1) {
    if (tid < st) { rs[tid] += rs[tid + st]; rq[tid] += rq[tid + st]; }
    __syncthreads();
  }
  const float mu = rs[0] * (1.f / EMB);
  const float var = rq[0] * (1.f / EMB) - mu * mu;
  const float rstd = rsqrtf(var + 1e-5f);
  bf16_t* orow = out + (size_t)row * EMB;
  const int c = tid * 4;
  orow[c + 0] = (bf16_t)((v.x - mu) * rstd * gw[c + 0] + gb[c + 0]);
  orow[c + 1] = (bf16_t)((v.y - mu) * rstd * gw[c + 1] + gb[c + 1]);
  orow[c + 2] = (bf16_t)((v.z - mu) * rstd * gw[c + 2] + gb[c + 2]);
  orow[c + 3] = (bf16_t)((v.w - mu) * rstd * gw[c + 3] + gb[c + 3]);
}

// ---------------------------------------------------------------- GEMM
// C[M,N] = A[M,K] @ Bw[N,K]^T (+epilogue). A,Bw row-major bf16, K contiguous.
// Block tile 128x128, 8 waves, wave tile 32x64 (2x4 WMMA accumulators).
// K staged in LDS (BK=64, double buffered) via global_load_async_to_lds_b128.
// EPI 0: outb = (acc+bias)*scale   (bf16)
// EPI 1: outf = acc+bias+res      (f32)
// EPI 2: outb = gelu(acc+bias)    (bf16)
#define BKG 64
template <int EPI>
__global__ __launch_bounds__(256) void gemm_bf16_k(const bf16_t* __restrict__ A,
                                                   const bf16_t* __restrict__ Bw,
                                                   const float* __restrict__ bias,
                                                   const float* __restrict__ res,
                                                   bf16_t* __restrict__ outb,
                                                   float* __restrict__ outf,
                                                   int M, int N, int K, float scale) {
  __shared__ __attribute__((aligned(32))) bf16_t As[2][128][72];
  __shared__ __attribute__((aligned(32))) bf16_t Bs[2][128][72];
  const int tid = threadIdx.x, lane = tid & 31, wv = tid >> 5;
  const int wm = (wv & 3) * 32, wn = (wv >> 2) * 64;
  const int bm = blockIdx.y * 128, bn = blockIdx.x * 128;
  const int hg = lane >> 4, l15 = lane & 15;

  v8f acc[2][4];
#pragma unroll
  for (int mt = 0; mt < 2; ++mt)
#pragma unroll
    for (int nt = 0; nt < 4; ++nt) acc[mt][nt] = v8f_zero();

  // 128x64 tile = 1024 16B chunks per array; 4 per thread
  auto issue_tile = [&](int k0, int buf) {
#pragma unroll
    for (int i = 0; i < 4; ++i) {
      const int c = tid + 256 * i, rr = c >> 3, ss = (c & 7) * 8;
      async_ld_b128((uint32_t)(uintptr_t)&As[buf][rr][ss],
                    A + (size_t)(bm + rr) * K + k0 + ss);
      async_ld_b128((uint32_t)(uintptr_t)&Bs[buf][rr][ss],
                    Bw + (size_t)(bn + rr) * K + k0 + ss);
    }
  };

  issue_tile(0, 0);
  wait_async0();
  __syncthreads();

  const int nk = K / BKG;
  for (int kk = 0; kk < nk; ++kk) {
    const int cur = kk & 1;
    if (kk + 1 < nk) issue_tile((kk + 1) * BKG, cur ^ 1);
    if (kk + 2 < nk) {
      __builtin_prefetch(A + (size_t)(bm + (tid >> 3)) * K + (kk + 2) * BKG, 0, 0);
      __builtin_prefetch(Bw + (size_t)(bn + (tid >> 3)) * K + (kk + 2) * BKG, 0, 0);
    }
#pragma unroll
    for (int ks = 0; ks < 2; ++ks) {
      Frag16 af[2];
#pragma unroll
      for (int mt = 0; mt < 2; ++mt) {
        const bf16_t* p = &As[cur][wm + mt * 16 + l15][ks * 32];
        af[mt].u[0] = *(const uint4*)(p + hg * 8);
        af[mt].u[1] = *(const uint4*)(p + 16 + hg * 8);
      }
#pragma unroll
      for (int nt = 0; nt < 4; ++nt) {
        Frag16 bfr;
        const bf16_t* p = &Bs[cur][wn + nt * 16 + l15][ks * 32];
        bfr.u[0] = *(const uint4*)(p + hg * 8);
        bfr.u[1] = *(const uint4*)(p + 16 + hg * 8);
#pragma unroll
        for (int mt = 0; mt < 2; ++mt)
          acc[mt][nt] = wmma_bf16(af[mt].v, bfr.v, acc[mt][nt]);
      }
    }
    if (kk + 1 < nk) {
      wait_async0();
      __syncthreads();
    }
  }

#pragma unroll
  for (int mt = 0; mt < 2; ++mt) {
#pragma unroll
    for (int nt = 0; nt < 4; ++nt) {
      const int col = bn + wn + nt * 16 + l15;
      const int row0 = bm + wm + mt * 16 + hg * 8;
      const float bb = bias[col];
#pragma unroll
      for (int r = 0; r < 8; ++r) {
        const size_t idx = (size_t)(row0 + r) * N + col;
        const float val = acc[mt][nt][r] + bb;
        if (EPI == 0) {
          outb[idx] = (bf16_t)(val * scale);
        } else if (EPI == 1) {
          outf[idx] = val + res[idx];
        } else {
          const float g =
              0.5f * val * (1.f + tanhf(0.7978845608028654f * (val + 0.044715f * val * val * val)));
          outb[idx] = (bf16_t)g;
        }
      }
    }
  }
}

// ---------------------------------------------------------------- Flash attention
// Q,K,V bf16 in [S, EMB] with head h at cols h*64..h*64+63 (Q pre-scaled 1/8).
// Block = (64 q-rows, one head), 4 waves, each wave owns 16 q-rows.
// K tile DMA'd to LDS async; V tile transposed into LDS with packed b32 stores.
__global__ __launch_bounds__(128) void flash_attn_k(const bf16_t* __restrict__ Q,
                                                    const bf16_t* __restrict__ Kb,
                                                    const bf16_t* __restrict__ Vb,
                                                    bf16_t* __restrict__ O) {
  __shared__ __attribute__((aligned(32))) bf16_t Kt[32][72];     // [kv][d]
  __shared__ __attribute__((aligned(32))) bf16_t Vt[64][40];     // [d][kv] (transposed)
  __shared__ __attribute__((aligned(32))) bf16_t Ps[4][16][40];  // per-wave P bounce
  const int h = blockIdx.y, qblk = blockIdx.x * 64;
  const int tid = threadIdx.x, wv = tid >> 5, lane = tid & 31;
  const int hg = lane >> 4, l15 = lane & 15;
  const int qrow0 = qblk + wv * 16;
  const int hc = h * HD;

  Frag16 qa[2];
#pragma unroll
  for (int ks = 0; ks < 2; ++ks) {
    const bf16_t* p = Q + (size_t)(qrow0 + l15) * EMB + hc + ks * 32 + hg * 8;
    qa[ks].u[0] = *(const uint4*)p;
    qa[ks].u[1] = *(const uint4*)(p + 16);
  }
  v8f o[4];
#pragma unroll
  for (int nt = 0; nt < 4; ++nt) o[nt] = v8f_zero();
  float mstat[8], lstat[8];
#pragma unroll
  for (int r = 0; r < 8; ++r) { mstat[r] = -1e30f; lstat[r] = 0.f; }

  const int kv_end = qblk + 64;
  for (int kv = 0; kv < kv_end; kv += 32) {
    __syncthreads();  // previous iteration's Kt/Vt reads complete
    {
      // K tile: 32x64, async DMA straight into LDS (2x b128 per thread)
      const int rr = tid >> 2, d0 = (tid & 3) * 16;
      const bf16_t* ksrc = Kb + (size_t)(kv + rr) * EMB + hc + d0;
      async_ld_b128((uint32_t)(uintptr_t)&Kt[rr][d0], ksrc);
      async_ld_b128((uint32_t)(uintptr_t)&Kt[rr][d0 + 8], ksrc + 8);
      // V tile transposed: two kv rows packed per b32 store
      const int p2 = (tid >> 3) * 2, dg = (tid & 7) * 8;
      union { uint4 q; unsigned short s[8]; } v0u, v1u;
      v0u.q = *(const uint4*)(Vb + (size_t)(kv + p2) * EMB + hc + dg);
      v1u.q = *(const uint4*)(Vb + (size_t)(kv + p2 + 1) * EMB + hc + dg);
#pragma unroll
      for (int j = 0; j < 8; ++j) {
        const unsigned int pk = (unsigned int)v0u.s[j] | ((unsigned int)v1u.s[j] << 16);
        *(unsigned int*)&Vt[dg + j][p2] = pk;
      }
    }
    wait_async0();
    __syncthreads();

    // scores: two 16x16 tiles, K-dim = 64 = 2 WMMA steps each
    v8f s0 = v8f_zero(), s1 = v8f_zero();
#pragma unroll
    for (int ks = 0; ks < 2; ++ks) {
      Frag16 b0, b1;
      const bf16_t* p0 = &Kt[l15][ks * 32 + hg * 8];
      b0.u[0] = *(const uint4*)p0; b0.u[1] = *(const uint4*)(p0 + 16);
      const bf16_t* p1 = &Kt[16 + l15][ks * 32 + hg * 8];
      b1.u[0] = *(const uint4*)p1; b1.u[1] = *(const uint4*)(p1 + 16);
      s0 = wmma_bf16(qa[ks].v, b0.v, s0);
      s1 = wmma_bf16(qa[ks].v, b1.v, s1);
    }

    // causal mask + online softmax (row m = hg*8+r, col = lane&15 within tile)
#pragma unroll
    for (int r = 0; r < 8; ++r) {
      const int rowg = qrow0 + hg * 8 + r;
      const float v0 = (kv + l15 <= rowg) ? s0[r] : -1e30f;
      const float v1 = (kv + 16 + l15 <= rowg) ? s1[r] : -1e30f;
      float mx = fmaxf(v0, v1);
#pragma unroll
      for (int d = 1; d < 16; d <<= 1) mx = fmaxf(mx, __shfl_xor(mx, d, 32));
      const float mnew = fmaxf(mstat[r], mx);
      const float alpha = __expf(mstat[r] - mnew);
      const float p0 = __expf(v0 - mnew);
      const float p1 = __expf(v1 - mnew);
      float rsum = p0 + p1;
#pragma unroll
      for (int d = 1; d < 16; d <<= 1) rsum += __shfl_xor(rsum, d, 32);
      lstat[r] = lstat[r] * alpha + rsum;
      mstat[r] = mnew;
#pragma unroll
      for (int nt = 0; nt < 4; ++nt) o[nt][r] *= alpha;
      Ps[wv][hg * 8 + r][l15] = (bf16_t)p0;
      Ps[wv][hg * 8 + r][16 + l15] = (bf16_t)p1;
    }
    __syncthreads();  // P visible; Vt still valid

    // PV: P(16x32) @ V(32x64) -> 4 WMMAs
    Frag16 pa;
    const bf16_t* pp = &Ps[wv][l15][0];
    pa.u[0] = *(const uint4*)(pp + hg * 8);
    pa.u[1] = *(const uint4*)(pp + 16 + hg * 8);
#pragma unroll
    for (int nt = 0; nt < 4; ++nt) {
      Frag16 vb;
      const bf16_t* p = &Vt[nt * 16 + l15][hg * 8];
      vb.u[0] = *(const uint4*)p;
      vb.u[1] = *(const uint4*)(p + 16);
      o[nt] = wmma_bf16(pa.v, vb.v, o[nt]);
    }
  }

#pragma unroll
  for (int r = 0; r < 8; ++r) {
    const float inv = 1.f / lstat[r];
    const size_t rowoff = (size_t)(qrow0 + hg * 8 + r) * EMB + hc;
#pragma unroll
    for (int nt = 0; nt < 4; ++nt)
      O[rowoff + nt * 16 + l15] = (bf16_t)(o[nt][r] * inv);
  }
}

// ---------------------------------------------------------------- launch
extern "C" void kernel_launch(void* const* d_in, const int* in_sizes, int n_in,
                              void* d_out, int out_size, void* d_ws, size_t ws_size,
                              hipStream_t stream) {
  (void)in_sizes; (void)n_in; (void)out_size; (void)ws_size;
  const float* x   = (const float*)d_in[0];
  const float* Wq  = (const float*)d_in[1];
  const float* bq  = (const float*)d_in[2];
  const float* Wk  = (const float*)d_in[3];
  const float* bk  = (const float*)d_in[4];
  const float* Wv  = (const float*)d_in[5];
  const float* bv  = (const float*)d_in[6];
  const float* Wo  = (const float*)d_in[7];
  const float* bo  = (const float*)d_in[8];
  const float* W1  = (const float*)d_in[9];
  const float* b1  = (const float*)d_in[10];
  const float* W2  = (const float*)d_in[11];
  const float* b2  = (const float*)d_in[12];
  const float* g1  = (const float*)d_in[13];
  const float* be1 = (const float*)d_in[14];
  const float* g2  = (const float*)d_in[15];
  const float* be2 = (const float*)d_in[16];

  char* ws = (char*)d_ws;
  size_t off = 0;
  auto alloc = [&](size_t bytes) -> char* {
    char* p = ws + off;
    off += (bytes + 255) & ~(size_t)255;
    return p;
  };
  bf16_t* WqB = (bf16_t*)alloc((size_t)EMB * EMB * 2);
  bf16_t* WkB = (bf16_t*)alloc((size_t)EMB * EMB * 2);
  bf16_t* WvB = (bf16_t*)alloc((size_t)EMB * EMB * 2);
  bf16_t* WoB = (bf16_t*)alloc((size_t)EMB * EMB * 2);
  bf16_t* W1B = (bf16_t*)alloc((size_t)DFF * EMB * 2);
  bf16_t* W2B = (bf16_t*)alloc((size_t)EMB * DFF * 2);
  bf16_t* hB  = (bf16_t*)alloc((size_t)SEQ * EMB * 2);
  bf16_t* qB  = (bf16_t*)alloc((size_t)SEQ * EMB * 2);
  bf16_t* kB  = (bf16_t*)alloc((size_t)SEQ * EMB * 2);
  bf16_t* vB  = (bf16_t*)alloc((size_t)SEQ * EMB * 2);
  bf16_t* cB  = (bf16_t*)alloc((size_t)SEQ * EMB * 2);
  float*  x1  = (float*)alloc((size_t)SEQ * EMB * 4);
  bf16_t* h2B = (bf16_t*)alloc((size_t)SEQ * EMB * 2);
  bf16_t* gB  = (bf16_t*)alloc((size_t)SEQ * DFF * 2);

  // weight conversion f32 -> bf16 (24 MB bf16: resident in 192 MB L2)
  cvt_f32_bf16<<<EMB * EMB / 1024, 256, 0, stream>>>(Wq, WqB, EMB * EMB);
  cvt_f32_bf16<<<EMB * EMB / 1024, 256, 0, stream>>>(Wk, WkB, EMB * EMB);
  cvt_f32_bf16<<<EMB * EMB / 1024, 256, 0, stream>>>(Wv, WvB, EMB * EMB);
  cvt_f32_bf16<<<EMB * EMB / 1024, 256, 0, stream>>>(Wo, WoB, EMB * EMB);
  cvt_f32_bf16<<<DFF * EMB / 1024, 256, 0, stream>>>(W1, W1B, DFF * EMB);
  cvt_f32_bf16<<<EMB * DFF / 1024, 256, 0, stream>>>(W2, W2B, EMB * DFF);

  const dim3 gEE(EMB / 128, SEQ / 128);

  // pre-LN 1 + QKV projections (Q pre-scaled by 1/sqrt(64))
  layernorm_bf16<<<SEQ, 256, 0, stream>>>(x, g1, be1, hB);
  gemm_bf16_k<0><<<gEE, 256, 0, stream>>>(hB, WqB, bq, nullptr, qB, nullptr, SEQ, EMB, EMB, 0.125f);
  gemm_bf16_k<0><<<gEE, 256, 0, stream>>>(hB, WkB, bk, nullptr, kB, nullptr, SEQ, EMB, EMB, 1.0f);
  gemm_bf16_k<0><<<gEE, 256, 0, stream>>>(hB, WvB, bv, nullptr, vB, nullptr, SEQ, EMB, EMB, 1.0f);

  // causal flash attention
  flash_attn_k<<<dim3(SEQ / 64, HEADS), 128, 0, stream>>>(qB, kB, vB, cB);

  // out-projection + residual -> x1 (f32)
  gemm_bf16_k<1><<<gEE, 256, 0, stream>>>(cB, WoB, bo, x, nullptr, x1, SEQ, EMB, EMB, 1.0f);

  // pre-LN 2 + MLP (GELU) + residual -> d_out (f32)
  layernorm_bf16<<<SEQ, 256, 0, stream>>>(x1, g2, be2, h2B);
  gemm_bf16_k<2><<<dim3(DFF / 128, SEQ / 128), 256, 0, stream>>>(h2B, W1B, b1, nullptr, gB, nullptr,
                                                                 SEQ, DFF, EMB, 1.0f);
  gemm_bf16_k<1><<<gEE, 256, 0, stream>>>(gB, W2B, b2, x1, nullptr, (float*)d_out, SEQ, EMB, DFF,
                                          1.0f);
}
